// MultiheadBlockAttention_4715874091577
// MI455X (gfx1250) — matrix operand order repaired and checked by reading, not compiled, not run on hardware
//
#include <hip/hip_runtime.h>
#include <stdint.h>

// ---------------- problem constants ----------------
constexpr int Bc = 8, Tt = 513, Mm = 8, Dd = 512, Oo = 64, Hh = 8, DH = 64, Ll = 512, Sx = 72;
constexpr int ROWS_T   = Bc * Ll * Mm;        // 32768 temporal rows
constexpr int ROWS_ALL = ROWS_T + Bc * Sx;    // 33344 (+ ss rows)
constexpr int ROWS_PAD = 33408;               // 261 * 128 (zero-padded rows)
constexpr size_t OFF_OTHERS = (size_t)Bc * Tt * Mm * Dd;                 // 16,809,984
constexpr size_t OFF_TATTN  = OFF_OTHERS + (size_t)Bc * Oo * Dd;         // +262,144
constexpr size_t OFF_OATTN  = OFF_TATTN + (size_t)Bc * Hh * Ll * Mm * Sx;// +18,874,368

// ---------------- WMMA types / helpers ----------------
typedef __attribute__((ext_vector_type(16))) __bf16 v16bf;
typedef __attribute__((ext_vector_type(8)))  float  v8f;

union Frag { v16bf v; uint4 q[2]; };

__device__ __forceinline__ unsigned short f2bf(float f) {
  unsigned int u = __float_as_uint(f);
  u += 0x7FFFu + ((u >> 16) & 1u);                 // round-to-nearest-even
  return (unsigned short)(u >> 16);
}
__device__ __forceinline__ float bf2f(unsigned short s) {
  return __uint_as_float(((unsigned int)s) << 16);
}

#define WMMA_BF16(A, Bv, C) \
  __builtin_amdgcn_wmma_f32_16x16x32_bf16(false, (A), false, (Bv), (short)0, (C), false, false)

// CDNA5 async direct-to-LDS copy (ASYNCcnt-tracked), GVS addressing:
//   mem = SGPR_base + VGPR_i32_offset ; LDS address taken from the vdst VGPR
__device__ __forceinline__ void async_ld16(const unsigned short* base, unsigned byte_off, void* ldsp) {
  unsigned lds = (unsigned)(uintptr_t)ldsp;        // flat->LDS: low 32 bits
  asm volatile("global_load_async_to_lds_b128 %0, %1, %2"
               :: "v"(lds), "v"(byte_off), "s"(base) : "memory");
}
__device__ __forceinline__ void wait_async0() {
  asm volatile("s_wait_asynccnt 0x0" ::: "memory");
}

// A-fragment (16x32 bf16): lane holds row (lane&15); 16 contiguous halves at base+hi*16.
// B-fragment via "transposed storage": lane holds col n=(lane&15); 16 contiguous halves at stored[n][khalf*16].

// ---------------- kernel 1: fp32 -> bf16 conversion / packing ----------------
__global__ __launch_bounds__(256) void k_convert(
    const float* __restrict__ temporal, const float* __restrict__ others,
    const float* __restrict__ Wq, const float* __restrict__ Wk, const float* __restrict__ Wv,
    unsigned short* __restrict__ xall, unsigned short* __restrict__ wbf)
{
  const int blk = blockIdx.x, t = threadIdx.x;
  const float* src = nullptr;
  unsigned short* dst;
  if (blk < ROWS_PAD) {
    const int row = blk;
    if (row < ROWS_T) {                       // temp = temporal[:,1:]
      int b = row >> 12, rem = row & 4095, l = rem >> 3, m = rem & 7;
      src = temporal + (((size_t)b * Tt + (l + 1)) * Mm + m) * Dd;
    } else if (row < ROWS_ALL) {              // ss = [temporal[:,0], others]
      int r2 = row - ROWS_T, b = r2 / Sx, s = r2 - b * Sx;
      if (s < Mm) src = temporal + (((size_t)b * Tt) * Mm + s) * Dd;
      else        src = others + ((size_t)b * Oo + (s - Mm)) * Dd;
    }                                          // else: zero pad row
    dst = xall + (size_t)row * Dd;
  } else {                                     // weights, row-major (j,d) as-is
    int blk2 = blk - ROWS_PAD;
    int mat = blk2 >> 9, j = blk2 & 511;
    src = (mat == 0 ? Wq : (mat == 1 ? Wk : Wv)) + (size_t)j * Dd;
    dst = wbf + (size_t)mat * Dd * Dd + (size_t)j * Dd;
  }
  unsigned int val = 0;
  if (src) {
    float2 f = *(const float2*)(src + 2 * t);
    val = (unsigned int)f2bf(f.x) | ((unsigned int)f2bf(f.y) << 16);
  }
  ((unsigned int*)dst)[t] = val;
}

// ---------------- kernel 2: QKV projection GEMM (C = X @ W^T + b), bf16 WMMA ----------------
// grid: (261 row-tiles of 128, 8 col-tiles of 64, 3 matrices); 256 threads = 8 waves (4x2),
// each wave owns a 32x32 output tile. K loop: 512 in steps of 32, double-buffered LDS filled
// with GLOBAL_LOAD_ASYNC_TO_LDS_B128 (loads for step+1 in flight while step's WMMAs run).
__global__ __launch_bounds__(256) void k_gemm(
    const unsigned short* __restrict__ xall, const unsigned short* __restrict__ wbf,
    const float* __restrict__ bq, const float* __restrict__ bk, const float* __restrict__ bv,
    unsigned short* __restrict__ qb, unsigned short* __restrict__ kb, unsigned short* __restrict__ vb)
{
  const int rt = blockIdx.x, nt = blockIdx.y, mat = blockIdx.z;
  const unsigned short* w = wbf + (size_t)mat * Dd * Dd;
  const float* bias = (mat == 0 ? bq : (mat == 1 ? bk : bv));
  unsigned short* outp = (mat == 0 ? qb : (mat == 1 ? kb : vb));

  __shared__ unsigned short As[2][128 * 40];  // 128 rows x 32 k, stride 40 (16B-aligned, bank pad)
  __shared__ unsigned short Bs[2][64 * 40];   // 64 cols(j) x 32 k

  const int t = threadIdx.x, wvi = t >> 5, lane = t & 31;
  const int wm = wvi & 3, wn = wvi >> 2;
  const int rr = lane & 15, hi = lane >> 4;
  const int row0 = rt * 128 + wm * 32;
  const int col0 = nt * 64 + wn * 32;

  // staging coords: 16B chunks (8 halves). A: 512 chunks -> 2/thread; B: 256 chunks -> 1/thread.
  const int ar = t >> 2, ao = (t & 3) * 8;    // A rows ar and ar+64
  const int br = t >> 2, bo = (t & 3) * 8;

  auto issue = [&](int buf, int k0) {
    async_ld16(xall, (unsigned)((((rt * 128 + ar) * Dd) + k0 + ao) * 2),        &As[buf][ar * 40 + ao]);
    async_ld16(xall, (unsigned)((((rt * 128 + ar + 64) * Dd) + k0 + ao) * 2),   &As[buf][(ar + 64) * 40 + ao]);
    async_ld16(w,    (unsigned)((((nt * 64 + br) * Dd) + k0 + bo) * 2),         &Bs[buf][br * 40 + bo]);
  };

  issue(0, 0);
  v8f acc[2][2] = {};
  for (int step = 0; step < 16; step++) {
    const int buf = step & 1;
    wait_async0();            // own async loads into buf complete
    __syncthreads();          // everyone's loads complete; prev readers of buf^1 are done
    if (step < 15) issue(buf ^ 1, (step + 1) * 32);
    v16bf a[2], bfr[2];
    #pragma unroll
    for (int i = 0; i < 2; i++) {
      Frag f; const unsigned short* p = &As[buf][(wm * 32 + i * 16 + rr) * 40 + hi * 16];
      f.q[0] = *(const uint4*)p; f.q[1] = *(const uint4*)(p + 8); a[i] = f.v;
    }
    #pragma unroll
    for (int j = 0; j < 2; j++) {
      Frag f; const unsigned short* p = &Bs[buf][(wn * 32 + j * 16 + rr) * 40 + hi * 16];
      f.q[0] = *(const uint4*)p; f.q[1] = *(const uint4*)(p + 8); bfr[j] = f.v;
    }
    #pragma unroll
    for (int i = 0; i < 2; i++)
      #pragma unroll
      for (int j = 0; j < 2; j++)
        acc[i][j] = WMMA_BF16(a[i], bfr[j], acc[i][j]);
  }
  float bj[2];
  #pragma unroll
  for (int j = 0; j < 2; j++) bj[j] = bias[col0 + j * 16 + rr];
  #pragma unroll
  for (int i = 0; i < 2; i++)
    #pragma unroll
    for (int j = 0; j < 2; j++)
      #pragma unroll
      for (int r = 0; r < 8; r++)
        outp[((size_t)(row0 + i * 16 + hi * 8 + r)) * Dd + col0 + j * 16 + rr] =
            f2bf(acc[i][j][r] + bj[j]);
}

// ---------------- kernel 3: build V^T (d-major per (b,h)) for P@V fragments ----------------
// grid: (64 bh, 66 tiles): tiles 0..63 -> VT_tt (4096 tokens), 64..65 -> VT_ss (96 tokens, zero-padded)
__global__ __launch_bounds__(256) void k_transpose(
    const unsigned short* __restrict__ vbuf,
    unsigned short* __restrict__ vt_tt, unsigned short* __restrict__ vt_ss)
{
  const int bh = blockIdx.x, b = bh >> 3;
  const int tile = blockIdx.y;
  __shared__ unsigned short tb[64 * 72];      // stride 72 halves = 144B (16B aligned)
  const int t = threadIdx.x, h = bh & 7;
  if (tile < 64) {
    const int tok0 = tile * 64;
    { int r = t >> 2, o = (t & 3) * 16;
      const unsigned short* src = &vbuf[((size_t)(b * 4096 + tok0 + r)) * Dd + h * DH + o];
      *(uint4*)&tb[r * 72 + o]     = *(const uint4*)src;
      *(uint4*)&tb[r * 72 + o + 8] = *(const uint4*)(src + 8); }
    __syncthreads();
    { int d = t >> 2, o = (t & 3) * 16;
      union { unsigned short s[16]; uint4 q[2]; } u;
      #pragma unroll
      for (int i = 0; i < 16; i++) u.s[i] = tb[(o + i) * 72 + d];
      uint4* dst = (uint4*)&vt_tt[((size_t)bh * DH + d) * 4096 + tok0 + o];
      dst[0] = u.q[0]; dst[1] = u.q[1]; }
  } else {
    const int tok0 = (tile - 64) * 64;        // 0 or 64
    { int r = t >> 2, o = (t & 3) * 16;
      int tok = tok0 + r;
      uint4 v0 = make_uint4(0u, 0u, 0u, 0u), v1 = make_uint4(0u, 0u, 0u, 0u);
      if (tok < Sx) {
        const unsigned short* src = &vbuf[((size_t)(ROWS_T + b * Sx + tok)) * Dd + h * DH + o];
        v0 = *(const uint4*)src; v1 = *(const uint4*)(src + 8);
      }
      *(uint4*)&tb[r * 72 + o] = v0;
      *(uint4*)&tb[r * 72 + o + 8] = v1; }
    __syncthreads();
    { int d = t >> 2, o = (t & 3) * 16;
      if (tok0 + o < 96) {
        union { unsigned short s[16]; uint4 q[2]; } u;
        #pragma unroll
        for (int i = 0; i < 16; i++) u.s[i] = tb[(o + i) * 72 + d];
        uint4* dst = (uint4*)&vt_ss[((size_t)bh * DH + d) * 96 + tok0 + o];
        dst[0] = u.q[0]; dst[1] = u.q[1];
      } }
  }
}

// ---------------- kernel 4: temporal attention (WMMA, l-pairs per wave) ----------------
// grid: (8 l-chunks of 64, H, B). Each wave handles 4 l-pairs: 16 Q rows = {l0 m0..7, l1 m0..7}.
__global__ __launch_bounds__(256) void k_tattn(
    const unsigned short* __restrict__ qb, const unsigned short* __restrict__ kb,
    const unsigned short* __restrict__ vtt, const unsigned short* __restrict__ vtss,
    float* __restrict__ out)
{
  const int b = blockIdx.z, h = blockIdx.y;
  const int lbase = blockIdx.x * 64;
  const size_t bh = (size_t)b * Hh + h;
  __shared__ unsigned short KtsL[64 * 72];    // Kts row-major (o, d)
  __shared__ unsigned short VtsL[64 * 72];    // Vts^T (d, token)
  __shared__ unsigned short Pl[8 * 16 * 104]; // per-wave probs, 16 rows x 96 (pad 104), cols 80..95 stay 0
  const int t = threadIdx.x, wvi = t >> 5, lane = t & 31;
  const int rr = lane & 15, hi = lane >> 4;
  { int r = t >> 2, o = (t & 3) * 16;
    const unsigned short* ks = &kb[((size_t)(ROWS_T + b * Sx + Mm + r)) * Dd + h * DH + o];
    *(uint4*)&KtsL[r * 72 + o]     = *(const uint4*)ks;
    *(uint4*)&KtsL[r * 72 + o + 8] = *(const uint4*)(ks + 8);
    const unsigned short* vs = &vtss[((size_t)bh * DH + r) * 96 + Mm + o];
    *(uint4*)&VtsL[r * 72 + o]     = *(const uint4*)vs;
    *(uint4*)&VtsL[r * 72 + o + 8] = *(const uint4*)(vs + 8); }
  for (int i = t; i < 8 * 16 * 104 / 2; i += 256) ((unsigned int*)Pl)[i] = 0u;
  __syncthreads();

  unsigned short* pb = &Pl[wvi * (16 * 104)];
  const float scale = 0.125f;                 // 1/sqrt(64)
  const bool vtt_valid = ((rr < 8) == (hi == 0));

  for (int it = 0; it < 4; it++) {
    const int l0 = lbase + 2 * (wvi + 8 * it);
    const int lrow = l0 + hi;
    // Q A-fragments (rows are (l0,m) then (l1,m); tokens contiguous since l1=l0+1)
    v16bf aq[2];
    { const unsigned short* qrow = qb + ((size_t)(b * 4096 + l0 * 8 + rr)) * Dd + h * DH;
      #pragma unroll
      for (int kt = 0; kt < 2; kt++) {
        Frag f; const unsigned short* p = qrow + kt * 32 + hi * 16;
        f.q[0] = *(const uint4*)p; f.q[1] = *(const uint4*)(p + 8); aq[kt] = f.v; } }
    // tt scores: block-diagonal 16x16 (cols = K rows of l0|l1)
    v8f stt = {};
    { const unsigned short* krow = kb + ((size_t)(b * 4096 + l0 * 8 + rr)) * Dd + h * DH;
      #pragma unroll
      for (int kt = 0; kt < 2; kt++) {
        Frag f; const unsigned short* p = krow + kt * 32 + hi * 16;
        f.q[0] = *(const uint4*)p; f.q[1] = *(const uint4*)(p + 8);
        stt = WMMA_BF16(aq[kt], f.v, stt); } }
    // to scores: Q @ Kts^T (Kts shared in LDS)
    v8f sto[4] = {};
    #pragma unroll
    for (int ntl = 0; ntl < 4; ntl++)
      #pragma unroll
      for (int kt = 0; kt < 2; kt++) {
        Frag f; const unsigned short* p = &KtsL[(ntl * 16 + rr) * 72 + kt * 32 + hi * 16];
        f.q[0] = *(const uint4*)p; f.q[1] = *(const uint4*)(p + 8);
        sto[ntl] = WMMA_BF16(aq[kt], f.v, sto[ntl]);
      }
    // softmax per row (cols live across 16 lanes), write attn probs + P to LDS (bf16)
    #pragma unroll
    for (int r = 0; r < 8; r++) {
      float s_tt = vtt_valid ? stt[r] * scale : -1e30f;
      float s0 = sto[0][r] * scale, s1 = sto[1][r] * scale;
      float s2 = sto[2][r] * scale, s3 = sto[3][r] * scale;
      float mx = fmaxf(fmaxf(fmaxf(s0, s1), fmaxf(s2, s3)), s_tt);
      #pragma unroll
      for (int off = 8; off >= 1; off >>= 1) mx = fmaxf(mx, __shfl_xor(mx, off, 32));
      float ett = __expf(s_tt - mx);
      float e0 = __expf(s0 - mx), e1 = __expf(s1 - mx);
      float e2 = __expf(s2 - mx), e3 = __expf(s3 - mx);
      float sm = ett + e0 + e1 + e2 + e3;
      #pragma unroll
      for (int off = 8; off >= 1; off >>= 1) sm += __shfl_xor(sm, off, 32);
      float inv = 1.0f / sm;
      float ptt = ett * inv, p0 = e0 * inv, p1 = e1 * inv, p2 = e2 * inv, p3 = e3 * inv;
      size_t abase = OFF_TATTN + (((bh * Ll + lrow) * Mm + r) * (size_t)Sx);
      if (vtt_valid) out[abase + (rr & 7)] = ptt;      // [..., :M] block
      out[abase + 8 + 0  + rr] = p0;                   // [..., M:] blocks
      out[abase + 8 + 16 + rr] = p1;
      out[abase + 8 + 32 + rr] = p2;
      out[abase + 8 + 48 + rr] = p3;
      unsigned short* pr = pb + (r + 8 * hi) * 104;
      pr[rr] = f2bf(p0); pr[16 + rr] = f2bf(p1); pr[32 + rr] = f2bf(p2); pr[48 + rr] = f2bf(p3);
      pr[64 + rr] = f2bf(ptt);                          // invalid lanes write ~0 into mirror slots
    }
    // P @ [Vts ; Vtt(l0)|Vtt(l1) ; 0]  (K = 96)
    v16bf ap[3];
    #pragma unroll
    for (int kt = 0; kt < 3; kt++) {
      Frag f; const unsigned short* p = pb + rr * 104 + kt * 32 + hi * 16;
      f.q[0] = *(const uint4*)p; f.q[1] = *(const uint4*)(p + 8); ap[kt] = f.v;
    }
    v8f oacc[4] = {};
    #pragma unroll
    for (int ntl = 0; ntl < 4; ntl++) {
      #pragma unroll
      for (int kt = 0; kt < 2; kt++) {
        Frag f; const unsigned short* p = &VtsL[(ntl * 16 + rr) * 72 + kt * 32 + hi * 16];
        f.q[0] = *(const uint4*)p; f.q[1] = *(const uint4*)(p + 8);
        oacc[ntl] = WMMA_BF16(ap[kt], f.v, oacc[ntl]);
      }
      Frag f2;
      if (hi == 0) {  // k 0..15: tokens l0*8..l0*8+15 of Vtt^T (contiguous since l1=l0+1)
        const unsigned short* p = vtt + ((size_t)bh * DH + ntl * 16 + rr) * 4096 + l0 * 8;
        f2.q[0] = *(const uint4*)p; f2.q[1] = *(const uint4*)(p + 8);
      } else {        // k 16..31: zero pad
        f2.q[0] = make_uint4(0u, 0u, 0u, 0u); f2.q[1] = make_uint4(0u, 0u, 0u, 0u);
      }
      oacc[ntl] = WMMA_BF16(ap[2], f2.v, oacc[ntl]);
    }
    #pragma unroll
    for (int ntl = 0; ntl < 4; ntl++)
      #pragma unroll
      for (int r = 0; r < 8; r++)
        out[(((size_t)b * Tt + (1 + lrow)) * Mm + r) * Dd + h * DH + ntl * 16 + rr] = oacc[ntl][r];
  }
}

// ---------------- kernel 5: "others" attention (72 tokens, f32 VALU) ----------------
__global__ __launch_bounds__(256) void k_oattn(
    const unsigned short* __restrict__ qb, const unsigned short* __restrict__ kb,
    const unsigned short* __restrict__ vb, float* __restrict__ out)
{
  const int bh = blockIdx.x, b = bh >> 3, h = bh & 7;
  __shared__ unsigned short Qs[Sx * DH], Ks[Sx * DH], Vs[Sx * DH];
  __shared__ float Ssc[Sx * 76];
  const int t = threadIdx.x;
  const size_t rowbase = ROWS_T + (size_t)b * Sx;
  for (int i = t; i < Sx * DH; i += 256) {
    int tok = i >> 6, d = i & 63;
    size_t gi = (rowbase + tok) * (size_t)Dd + h * DH + d;
    Qs[i] = qb[gi]; Ks[i] = kb[gi]; Vs[i] = vb[gi];
  }
  __syncthreads();
  for (int i = t; i < Sx * Sx; i += 256) {
    int rq = i / Sx, ck = i - rq * Sx;
    float s = 0.f;
    for (int d = 0; d < DH; d++) s += bf2f(Qs[rq * DH + d]) * bf2f(Ks[ck * DH + d]);
    Ssc[rq * 76 + ck] = s * 0.125f;
  }
  __syncthreads();
  if (t < Sx) {
    float mx = -1e30f;
    for (int c = 0; c < Sx; c++) mx = fmaxf(mx, Ssc[t * 76 + c]);
    float sm = 0.f;
    for (int c = 0; c < Sx; c++) { float e = __expf(Ssc[t * 76 + c] - mx); Ssc[t * 76 + c] = e; sm += e; }
    float inv = 1.0f / sm;
    for (int c = 0; c < Sx; c++) {
      float p = Ssc[t * 76 + c] * inv; Ssc[t * 76 + c] = p;
      out[OFF_OATTN + (size_t)bh * (Sx * Sx) + t * Sx + c] = p;
    }
  }
  __syncthreads();
  for (int i = t; i < Sx * DH; i += 256) {
    int tok = i >> 6, d = i & 63;
    float s = 0.f;
    for (int c = 0; c < Sx; c++) s += Ssc[tok * 76 + c] * bf2f(Vs[c * DH + d]);
    size_t dst;
    if (tok < Mm) dst = (((size_t)b * Tt + 0) * Mm + tok) * Dd + h * DH + d;   // temporal_out[:,0]
    else          dst = OFF_OTHERS + ((size_t)b * Oo + (tok - Mm)) * Dd + h * DH + d;
    out[dst] = s;
  }
}

// ---------------- launch ----------------
extern "C" void kernel_launch(void* const* d_in, const int* in_sizes, int n_in,
                              void* d_out, int out_size, void* d_ws, size_t ws_size,
                              hipStream_t stream) {
  const float* temporal = (const float*)d_in[0];
  const float* others   = (const float*)d_in[1];
  const float* Wq = (const float*)d_in[2]; const float* bq = (const float*)d_in[3];
  const float* Wk = (const float*)d_in[4]; const float* bk = (const float*)d_in[5];
  const float* Wv = (const float*)d_in[6]; const float* bv = (const float*)d_in[7];

  unsigned short* ws   = (unsigned short*)d_ws;
  unsigned short* xall = ws;                                           // 33408*512
  unsigned short* wbf  = xall + (size_t)ROWS_PAD * Dd;                 // 3*512*512
  unsigned short* qbuf = wbf + (size_t)3 * Dd * Dd;
  unsigned short* kbuf = qbuf + (size_t)ROWS_PAD * Dd;
  unsigned short* vbuf = kbuf + (size_t)ROWS_PAD * Dd;
  unsigned short* vt_tt = vbuf + (size_t)ROWS_PAD * Dd;                // B*H*64*4096
  unsigned short* vt_ss = vt_tt + (size_t)Bc * Hh * DH * 4096;         // B*H*64*96
  float* outp = (float*)d_out;

  k_convert<<<ROWS_PAD + 3 * 512, 256, 0, stream>>>(temporal, others, Wq, Wk, Wv, xall, wbf);
  k_gemm<<<dim3(ROWS_PAD / 128, Dd / 64, 3), 256, 0, stream>>>(xall, wbf, bq, bk, bv, qbuf, kbuf, vbuf);
  k_transpose<<<dim3(Bc * Hh, 66), 256, 0, stream>>>(vbuf, vt_tt, vt_ss);
  k_tattn<<<dim3(Ll / 64, Hh, Bc), 256, 0, stream>>>(qbuf, kbuf, vt_tt, vt_ss, outp);
  k_oattn<<<Bc * Hh, 256, 0, stream>>>(qbuf, kbuf, vbuf, outp);
}
